// PaperInformedMambaBlock_23785528885643
// MI455X (gfx1250) — matrix operand order repaired and checked by reading, not compiled, not run on hardware
//
#include <hip/hip_runtime.h>
#include <hip/hip_bf16.h>
#include <cstddef>

// ---------------------------------------------------------------------------
// MI455X (gfx1250) Mamba block.
//   GEMMs: bf16 V_WMMA_F32_16X16X32_BF16, f32 accumulate. 32x32 wave tile
//   (4 accumulators / 4 WMMAs per K-step from 2 A-frags x 2 B-frags).
//   Fragment-major LDS (B transposed), every fragment = 2 ds_load_b128;
//   double-buffered LDS, software-pipelined global loads, compile-time N/K
//   so all staging loads use immediate offsets off one running pointer.
//   Scan: one wave32 per (b,d) channel, lane n = state n (N == 32 == wave).
// ---------------------------------------------------------------------------

typedef __attribute__((ext_vector_type(16))) __bf16 v16bf;
typedef __attribute__((ext_vector_type(8)))  __bf16 v8bf;
typedef __attribute__((ext_vector_type(8)))  float  v8f;

#define D_MODEL 1024
#define D_STATE 32
#define B_SZ    2
#define L_SEQ   1024
#define KCONV   4

#define BM 128
#define BN 64
#define BK 32
#define ROWE 40          // padded LDS row: 40 bf16 = 80 B (16B-aligned, no bank conflicts)

__device__ __forceinline__ float sigmoid_f(float v) {
    return 1.0f / (1.0f + __expf(-v));
}

__device__ __forceinline__ float softplus_clip(float v) {
    float sp = (v > 20.0f) ? v : log1pf(__expf(v));
    return fminf(fmaxf(sp, 0.001f), 0.2f);
}

__device__ __forceinline__ v16bf cat8(v8bf lo, v8bf hi) {
    return __builtin_shufflevector(lo, hi, 0, 1, 2, 3, 4, 5, 6, 7,
                                           8, 9, 10, 11, 12, 13, 14, 15);
}

__device__ __forceinline__ v16bf frag16(const __bf16* p0, const __bf16* p1) {
    return cat8(*(const v8bf*)p0, *(const v8bf*)p1);
}

// ---------------------------------------------------------------------------
// C = act(A @ W + bias).  A: MxK f32 row-major, W: KxN f32 row-major.
// 256 threads = 8 waves; 128x64 block tile; wave (wmo,wn) in 4x2 grid; each
// wave owns a 32x32 strip: acc[2][2], 2 A-frags x 2 B-frags -> 4 WMMAs/step.
// N, K compile-time so staging loads use immediate offsets.
// ACT: 0 = none, 1 = softplus+clip (dt path).
// ---------------------------------------------------------------------------
template <int ACT, int N, int K>
__global__ __launch_bounds__(256)
void gemm_bf16_wmma(const float* __restrict__ A, const float* __restrict__ W,
                    const float* __restrict__ bias, float* __restrict__ C,
                    int M) {
    // Double-buffered, fragment-major LDS tiles (30 KB total).
    __shared__ alignas(16) __bf16 As [2][BM][ROWE];  // A row-major  [m][k]
    __shared__ alignas(16) __bf16 Bst[2][BN][ROWE];  // B TRANSPOSED [n][k]

    const int tid  = threadIdx.x;
    const int wave = tid >> 5;
    const int lane = tid & 31;
    const int bm   = blockIdx.y * BM;
    const int bn   = blockIdx.x * BN;
    const int wmo  = (wave >> 1) * 32;   // 0,32,64,96
    const int wn   = (wave & 1) * 32;    // 0,32

    // Staging: A -> thread owns rows ar and ar+64, 8 consecutive k each.
    //          B -> thread owns column bcol, 8 consecutive k
    //               (coalesced 128B across a wave per k row).
    const int ar   = tid >> 2;           // 0..63
    const int ac   = (tid & 3) * 8;      // 0,8,16,24
    const int bcol = tid & 63;           // 0..63
    const int bk   = (tid >> 6) * 8;     // 0,8,16,24

    v8f acc00 = {}, acc01 = {}, acc10 = {}, acc11 = {};
    float aR[16], bR[8];

    const float* __restrict__ aPtr = A + (size_t)(bm + ar) * K + ac;
    const float* __restrict__ bPtr = W + (size_t)bk * N + bn + bcol;

    auto load_tile = [&]() {
        #pragma unroll
        for (int e = 0; e < 8; ++e) aR[e]     = aPtr[e];            // imm offs
        #pragma unroll
        for (int e = 0; e < 8; ++e) aR[8 + e] = aPtr[64 * K + e];   // imm offs
        #pragma unroll
        for (int e = 0; e < 8; ++e) bR[e]     = bPtr[e * N];        // imm offs
    };
    auto store_tile = [&](int buf) {
        #pragma unroll
        for (int e = 0; e < 8; ++e) As [buf][ar]     [ac + e] = (__bf16)aR[e];
        #pragma unroll
        for (int e = 0; e < 8; ++e) As [buf][ar + 64][ac + e] = (__bf16)aR[8 + e];
        #pragma unroll
        for (int e = 0; e < 8; ++e) Bst[buf][bcol]   [bk + e] = (__bf16)bR[e];
    };

    const int NT = K / BK;
    load_tile();

    for (int kt = 0; kt < NT; ++kt) {
        const int cur = kt & 1;
        store_tile(cur);
        __syncthreads();

        if (kt + 1 < NT) {
            aPtr += BK;                   // one pointer bump each
            bPtr += (size_t)BK * N;
            load_tile();                  // in flight over the WMMAs
            __builtin_prefetch(aPtr + BK, 0, 0);             // global_prefetch_b8
            __builtin_prefetch(bPtr + (size_t)BK * N, 0, 0);
        }

        // A fragments (16-bit 16x32 ISA layout): lane holds row (lane&15),
        // K chunks {kb..kb+7} and {16+kb..16+kb+7}, kb = (lane>>4)*8.
        const int kb = (lane >> 4) * 8;
        const __bf16* ar0 = &As[cur][wmo + (lane & 15)][0];
        const __bf16* ar1 = &As[cur][wmo + 16 + (lane & 15)][0];
        v16bf af0 = frag16(ar0 + kb, ar0 + 16 + kb);
        v16bf af1 = frag16(ar1 + kb, ar1 + 16 + kb);

        // B fragments (32x16 layout): lane holds column, K chunks
        // {ks..ks+7} and {16+ks..16+ks+7}, ks = (lane>>4)*8 (matches A).
        const int ks = (lane >> 4) * 8;
        const __bf16* b0 = &Bst[cur][wn + (lane & 15)][ks];
        const __bf16* b1 = &Bst[cur][wn + 16 + (lane & 15)][ks];
        v16bf bf0 = frag16(b0, b0 + 16);
        v16bf bf1 = frag16(b1, b1 + 16);

        acc00 = __builtin_amdgcn_wmma_f32_16x16x32_bf16(
            false, af0, false, bf0, (short)0, acc00, false, false);
        acc01 = __builtin_amdgcn_wmma_f32_16x16x32_bf16(
            false, af0, false, bf1, (short)0, acc01, false, false);
        acc10 = __builtin_amdgcn_wmma_f32_16x16x32_bf16(
            false, af1, false, bf0, (short)0, acc10, false, false);
        acc11 = __builtin_amdgcn_wmma_f32_16x16x32_bf16(
            false, af1, false, bf1, (short)0, acc11, false, false);
        // Single barrier per tile: the barrier at the top of iteration kt+1
        // orders these LDS reads before any overwrite of this buffer.
    }

    // Epilogue. f32 C/D layout: vgpr v -> row v (lanes 0-15) / v+8 (16-31).
    const int cn    = bn + wn + (lane & 15);
    const int rbase = bm + wmo + (lane >> 4) * 8;
    const float bia0 = bias[cn];
    const float bia1 = bias[cn + 16];
    #pragma unroll
    for (int v = 0; v < 8; ++v) {
        const int ra = rbase + v;        // af0 half (rows wmo..wmo+15)
        const int rb = rbase + 16 + v;   // af1 half (rows wmo+16..wmo+31)
        float o00 = acc00[v] + bia0;
        float o01 = acc01[v] + bia1;
        float o10 = acc10[v] + bia0;
        float o11 = acc11[v] + bia1;
        if (ACT == 1) {
            o00 = softplus_clip(o00); o01 = softplus_clip(o01);
            o10 = softplus_clip(o10); o11 = softplus_clip(o11);
        }
        C[(size_t)ra * N + cn]      = o00;
        C[(size_t)ra * N + cn + 16] = o01;
        C[(size_t)rb * N + cn]      = o10;
        C[(size_t)rb * N + cn + 16] = o11;
    }
}

// ---------------------------------------------------------------------------
// Causal depthwise conv (K=4) over L + SiLU.  xp = xz[:, :D].
// ---------------------------------------------------------------------------
__global__ __launch_bounds__(256)
void conv_silu_kernel(const float* __restrict__ xz,
                      const float* __restrict__ conv_w,
                      const float* __restrict__ conv_b,
                      float* __restrict__ xs) {
    const size_t i = (size_t)blockIdx.x * blockDim.x + threadIdx.x;
    const size_t total = (size_t)B_SZ * L_SEQ * D_MODEL;
    if (i >= total) return;
    const int d     = (int)(i % D_MODEL);
    const size_t bl = i / D_MODEL;          // b*L + l
    const int l     = (int)(bl % L_SEQ);

    float acc = conv_b[d];
    #pragma unroll
    for (int k = 0; k < KCONV; ++k) {
        const int ls = l - (KCONV - 1) + k;
        if (ls >= 0)
            acc += conv_w[d * KCONV + k] *
                   xz[(bl - (size_t)(KCONV - 1 - k)) * (2 * D_MODEL) + d];
    }
    xs[i] = acc * sigmoid_f(acc);
}

// ---------------------------------------------------------------------------
// Selective scan: wave32 per (b,d), lane n holds state n (N=32).
// Bt/Ct computed on the fly (materializing them would be 256 MB each).
// Chunked: each wave loads 32 timesteps at once (lane <-> timestep) and
// broadcasts via constant-lane shuffles; lane j keeps the reduced y of
// timestep j, so the sequential loop has no per-step loads or stores.
// Output fused with the SiLU(z) gate.
// ---------------------------------------------------------------------------
__global__ __launch_bounds__(256)
void scan_kernel(const float* __restrict__ xs, const float* __restrict__ dt,
                 const float* __restrict__ xz, const float* __restrict__ A_log,
                 const float* __restrict__ wB, const float* __restrict__ bB,
                 const float* __restrict__ wC, const float* __restrict__ bC,
                 float* __restrict__ yz) {
    const int wave = blockIdx.x * (blockDim.x >> 5) + (threadIdx.x >> 5);
    const int lane = threadIdx.x & 31;          // == state index n
    const int b = wave / D_MODEL;
    const int d = wave % D_MODEL;
    if (b >= B_SZ) return;

    const int dn = d * D_STATE + lane;
    const float a  = -__expf(fminf(fmaxf(A_log[dn], -5.0f), 2.0f));
    const float wb = wB[dn], bb = bB[dn];
    const float wc = wC[dn], bc = bC[dn];

    float st = 0.0f;
    for (int l0 = 0; l0 < L_SEQ; l0 += 32) {
        const size_t row = (size_t)b * L_SEQ + l0 + lane;
        const float xv_all  = xs[row * D_MODEL + d];
        const float dtv_all = dt[row * D_MODEL + d];
        const float zv      = xz[row * (2 * D_MODEL) + D_MODEL + d];
        const float gate    = zv * sigmoid_f(zv);

        float ymine = 0.0f;
        #pragma unroll
        for (int j = 0; j < 32; ++j) {
            const float xv  = __shfl(xv_all,  j, 32);
            const float dtv = __shfl(dtv_all, j, 32);
            // a <= 0, dt >= 0  ->  dt*a <= 0 already; clamp below at -10.
            const float abar = __expf(fmaxf(dtv * a, -10.0f));
            st = st * abar + dtv * (xv * wb + bb) * xv;
            st = fminf(fmaxf(st, -10.0f), 10.0f);
            float y = (xv * wc + bc) * st;
            #pragma unroll
            for (int off = 16; off; off >>= 1)
                y += __shfl_xor(y, off, 32);
            if (lane == j) ymine = y;           // lane j owns timestep l0+j
        }
        yz[row * D_MODEL + d] = ymine * gate;
    }
}

// ---------------------------------------------------------------------------
// h = out + residual; LayerNorm(h) * g + b.  One 256-thread block per row.
// ---------------------------------------------------------------------------
__global__ __launch_bounds__(256)
void add_layernorm_kernel(const float* __restrict__ out,
                          const float* __restrict__ x,
                          const float* __restrict__ g,
                          const float* __restrict__ beta,
                          float* __restrict__ o) {
    const int row = blockIdx.x;
    const float* po = out + (size_t)row * D_MODEL;
    const float* px = x   + (size_t)row * D_MODEL;
    float*       pr = o   + (size_t)row * D_MODEL;

    float h[4], s = 0.0f, ss = 0.0f;
    #pragma unroll
    for (int j = 0; j < 4; ++j) {
        const int c = threadIdx.x + j * 256;
        const float v = po[c] + px[c];
        h[j] = v; s += v; ss += v * v;
    }
    #pragma unroll
    for (int off = 16; off; off >>= 1) {
        s  += __shfl_xor(s,  off, 32);
        ss += __shfl_xor(ss, off, 32);
    }
    __shared__ float s_s[8], s_ss[8];
    const int wave = threadIdx.x >> 5, lane = threadIdx.x & 31;
    if (lane == 0) { s_s[wave] = s; s_ss[wave] = ss; }
    __syncthreads();
    if (wave == 0) {
        s  = (lane < 8) ? s_s[lane]  : 0.0f;
        ss = (lane < 8) ? s_ss[lane] : 0.0f;
        #pragma unroll
        for (int off = 4; off; off >>= 1) {
            s  += __shfl_xor(s,  off, 32);
            ss += __shfl_xor(ss, off, 32);
        }
        if (lane == 0) { s_s[0] = s; s_ss[0] = ss; }
    }
    __syncthreads();
    const float mu  = s_s[0] * (1.0f / D_MODEL);
    const float var = s_ss[0] * (1.0f / D_MODEL) - mu * mu;
    const float inv = rsqrtf(var + 1e-5f);
    #pragma unroll
    for (int j = 0; j < 4; ++j) {
        const int c = threadIdx.x + j * 256;
        pr[c] = (h[j] - mu) * inv * g[c] + beta[c];
    }
}

// ---------------------------------------------------------------------------
extern "C" void kernel_launch(void* const* d_in, const int* in_sizes, int n_in,
                              void* d_out, int out_size, void* d_ws, size_t ws_size,
                              hipStream_t stream) {
    const float* x      = (const float*)d_in[0];
    const float* W_in   = (const float*)d_in[1];
    const float* b_in   = (const float*)d_in[2];
    const float* conv_w = (const float*)d_in[3];
    const float* conv_b = (const float*)d_in[4];
    const float* A_log  = (const float*)d_in[5];
    const float* wB     = (const float*)d_in[6];
    const float* bB     = (const float*)d_in[7];
    const float* wC     = (const float*)d_in[8];
    const float* bC     = (const float*)d_in[9];
    const float* W_dt   = (const float*)d_in[10];
    const float* b_dt   = (const float*)d_in[11];
    const float* W_out  = (const float*)d_in[12];
    const float* b_out  = (const float*)d_in[13];
    const float* ln_g   = (const float*)d_in[14];
    const float* ln_b   = (const float*)d_in[15];
    float* out = (float*)d_out;

    const int M = B_SZ * L_SEQ;       // 2048
    const int D = D_MODEL;            // 1024

    // Workspace layout (floats): xz | xs | dt | yz | out_pre  (48 MB total)
    float* ws      = (float*)d_ws;
    float* xz_buf  = ws;                                  // M * 2D
    float* xs_buf  = xz_buf  + (size_t)M * 2 * D;         // M * D
    float* dt_buf  = xs_buf  + (size_t)M * D;             // M * D
    float* yz_buf  = dt_buf  + (size_t)M * D;             // M * D
    float* out_buf = yz_buf  + (size_t)M * D;             // M * D

    // 1) xz = x @ W_in + b_in               (2048 x 2048, K=1024)
    gemm_bf16_wmma<0, 2 * D_MODEL, D_MODEL>
        <<<dim3(2 * D / BN, M / BM), 256, 0, stream>>>(x, W_in, b_in, xz_buf, M);

    // 2) xs = silu(causal depthwise conv(xp))
    conv_silu_kernel<<<(M * D + 255) / 256, 256, 0, stream>>>(
        xz_buf, conv_w, conv_b, xs_buf);

    // 3) dt = clip(softplus(xs @ W_dt + b_dt))   (fused epilogue)
    gemm_bf16_wmma<1, D_MODEL, D_MODEL>
        <<<dim3(D / BN, M / BM), 256, 0, stream>>>(xs_buf, W_dt, b_dt, dt_buf, M);

    // 4) selective scan + SiLU(z) gate  (2048 waves, 8 per block)
    scan_kernel<<<(B_SZ * D_MODEL) / 8, 256, 0, stream>>>(
        xs_buf, dt_buf, xz_buf, A_log, wB, bB, wC, bC, yz_buf);

    // 5) out_pre = yz @ W_out + b_out
    gemm_bf16_wmma<0, D_MODEL, D_MODEL>
        <<<dim3(D / BN, M / BM), 256, 0, stream>>>(yz_buf, W_out, b_out, out_buf, M);

    // 6) h = out_pre + x; LayerNorm -> d_out
    add_layernorm_kernel<<<M, 256, 0, stream>>>(out_buf, x, ln_g, ln_b, out);
}
